// LightGCN_6064493822026
// MI455X (gfx1250) — matrix operand order repaired and checked by reading, not compiled, not run on hardware
//
#include <hip/hip_runtime.h>
#include <hip/hip_bf16.h>

#define NUM_USER  100000
#define NUM_ITEM  50000
#define DIM       64
#define NUM_LAYER 3
#define NUM_EDGES 2000000
#define BATCH     4096
#define NUM_NEG   50
#define LSCALE    0.25f   // 1/(NUM_LAYER+1)

typedef __attribute__((ext_vector_type(16))) _Float16 v16h;
typedef __attribute__((ext_vector_type(8)))  float    v8f;

// ---------------- degree + normalization ----------------
__global__ void deg_kernel(const int* __restrict__ eu, const int* __restrict__ ei,
                           float* __restrict__ du, float* __restrict__ di, int E) {
  int e = blockIdx.x * blockDim.x + threadIdx.x;
  if (e < E) {
    atomicAdd(&du[eu[e]], 1.0f);
    atomicAdd(&di[ei[e]], 1.0f);
  }
}

__global__ void vals_kernel(const int* __restrict__ eu, const int* __restrict__ ei,
                            const float* __restrict__ du, const float* __restrict__ di,
                            float* __restrict__ vals, int E) {
  int e = blockIdx.x * blockDim.x + threadIdx.x;
  if (e < E) {
    float a = fmaxf(du[eu[e]], 1.0f);
    float b = fmaxf(di[ei[e]], 1.0f);
    vals[e] = rsqrtf(a * b);
  }
}

// ---------------- edge-parallel SpMM scatter ----------------
// dst[sidx[e]] += vals[e] * src[gidx[e]]  (16 threads/edge, float4 each)
__global__ __launch_bounds__(256)
void spmm_scatter(const float* __restrict__ src, const int* __restrict__ gidx,
                  const int* __restrict__ sidx, const float* __restrict__ vals,
                  float* __restrict__ dst, int E) {
  int t = blockIdx.x * blockDim.x + threadIdx.x;
  int e = t >> 4;
  if (e >= E) return;
  int c = (t & 15) << 2;
  float v = vals[e];
  const float4 s = *reinterpret_cast<const float4*>(src + (size_t)gidx[e] * DIM + c);
  float* d = dst + (size_t)sidx[e] * DIM + c;
  atomicAdd(d + 0, s.x * v);
  atomicAdd(d + 1, s.y * v);
  atomicAdd(d + 2, s.z * v);
  atomicAdd(d + 3, s.w * v);
}

__global__ void vec_add(float* __restrict__ dst, const float* __restrict__ src, int n) {
  int i = blockIdx.x * blockDim.x + threadIdx.x;
  if (i < n) dst[i] += src[i];
}

// ---------------- WMMA scoring: one wave32 per batch user ----------------
// A (16x64, as 2x 16x32 f16) = 16 item rows (pos/neg slots, zero padded)
// B (64x16) = user vector broadcast into all 16 columns
// D[i][j] = item_i . user  (all columns equal) via v_wmma_f32_16x16x32_f16
__global__ __launch_bounds__(128)
void score_kernel(const float* __restrict__ usum, const float* __restrict__ isum,
                  const int* __restrict__ user, const int* __restrict__ item,
                  const int* __restrict__ negs,
                  float* __restrict__ pos_out, float* __restrict__ neg_out,
                  float* __restrict__ reg_out) {
  const int lane = threadIdx.x & 31;
  const int b = blockIdx.x * 4 + (threadIdx.x >> 5);   // grid sized exactly: no EXEC gaps

  const int hi  = (lane & 16) ? 1 : 0;
  const int row = lane & 15;

  // --- build B: user vector broadcast to all columns ---
  // B 32x16 f16 layout: lanes 0-15 hold K{0..7,16..23}, lanes 16-31 hold K{8..15,24..31}
  const float* urow = usum + (size_t)user[b] * DIM;
  v16h bm0, bm1;
  float acc2 = 0.0f;
#pragma unroll
  for (int j = 0; j < 16; ++j) {
    int k = (hi ? 8 : 0) + (j < 8 ? j : 8 + j);
    float v0 = urow[k]      * LSCALE;
    float v1 = urow[k + 32] * LSCALE;
    bm0[j] = (_Float16)v0;
    bm1[j] = (_Float16)v1;
    if (row == 0) acc2 += v0 * v0 + v1 * v1;  // lanes 0 & 16 jointly cover all 64 dims once
  }

  // --- 4 tiles of 16 item slots (slot 0 = pos, 1..50 = negs, >=51 padded zero) ---
  const int kbase = hi ? 8 : 0;   // A 16x32 f16 layout: lane<16 K{0..7,16..23}, lane>=16 K{8..15,24..31}
#pragma unroll
  for (int t = 0; t < 4; ++t) {
    int slot = t * 16 + row;
    bool valid = slot < 1 + NUM_NEG;
    int idx = 0;
    if (valid) idx = (slot == 0) ? item[b] : negs[b * NUM_NEG + slot - 1];
    const float* irow = isum + (size_t)idx * DIM;

    v16h a0, a1;
#pragma unroll
    for (int j = 0; j < 16; ++j) {
      int k = kbase + (j < 8 ? j : 8 + j);
      float v0 = valid ? irow[k]      * LSCALE : 0.0f;
      float v1 = valid ? irow[k + 32] * LSCALE : 0.0f;
      a0[j] = (_Float16)v0;
      a1[j] = (_Float16)v1;
      acc2 += v0 * v0 + v1 * v1;   // each (item,dim) element loaded exactly once per wave
    }

    v8f c = {};
    c = __builtin_amdgcn_wmma_f32_16x16x32_f16(false, a0, false, bm0, (short)0, c, false, false);
    c = __builtin_amdgcn_wmma_f32_16x16x32_f16(false, a1, false, bm1, (short)0, c, false, false);

    // C/D f32 layout: VGPR r -> M=r (lanes 0-15) / M=r+8 (lanes 16-31); take column 0 / 0'
    if (row == 0) {
      int sbase = t * 16 + (hi ? 8 : 0);
#pragma unroll
      for (int r = 0; r < 8; ++r) {
        int s = sbase + r;
        float sc = c[r];
        if (s == 0)                 pos_out[b] = sc;
        else if (s < 1 + NUM_NEG)   neg_out[b * NUM_NEG + s - 1] = sc;
      }
    }
  }

  // --- reg_loss partial: wave reduce + one atomic ---
#pragma unroll
  for (int off = 16; off > 0; off >>= 1) acc2 += __shfl_xor(acc2, off, 32);
  if (lane == 0) atomicAdd(reg_out, 0.5f * acc2 * (1.0f / (float)BATCH));
}

// ---------------- host orchestration ----------------
static inline size_t align64f(size_t n) { return (n + 63) & ~(size_t)63; }  // 256B align in floats

extern "C" void kernel_launch(void* const* d_in, const int* in_sizes, int n_in,
                              void* d_out, int out_size, void* d_ws, size_t ws_size,
                              hipStream_t stream) {
  (void)in_sizes; (void)n_in; (void)out_size; (void)ws_size;

  const float* user_tab = (const float*)d_in[0];
  const float* item_tab = (const float*)d_in[1];
  const int*   edge_u   = (const int*)d_in[2];
  const int*   edge_i   = (const int*)d_in[3];
  const int*   q_user   = (const int*)d_in[4];
  const int*   q_item   = (const int*)d_in[5];
  const int*   q_negs   = (const int*)d_in[6];
  float* out = (float*)d_out;

  float* pos_out = out;                              // [BATCH]
  float* neg_out = out + BATCH;                      // [BATCH*NUM_NEG]
  float* reg_out = out + BATCH + BATCH * NUM_NEG;    // [1]

  // workspace layout
  float* ws = (float*)d_ws;
  size_t o = 0;
  float* deg_u = ws + o; o += align64f(NUM_USER);
  float* deg_i = ws + o; o += align64f(NUM_ITEM);
  float* vals  = ws + o; o += align64f(NUM_EDGES);
  const size_t NU = (size_t)NUM_USER * DIM;   // 6.4M floats
  const size_t NI = (size_t)NUM_ITEM * DIM;   // 3.2M floats
  float* u_sum = ws + o; o += NU;
  float* ubuf0 = ws + o; o += NU;
  float* ubuf1 = ws + o; o += NU;
  float* i_sum = ws + o; o += NI;
  float* ibuf0 = ws + o; o += NI;
  float* ibuf1 = ws + o; o += NI;

  const int E = NUM_EDGES;

  // 1) degrees + symmetric-normalization values
  hipMemsetAsync(deg_u, 0, NUM_USER * sizeof(float), stream);
  hipMemsetAsync(deg_i, 0, NUM_ITEM * sizeof(float), stream);
  deg_kernel<<<(E + 255) / 256, 256, 0, stream>>>(edge_u, edge_i, deg_u, deg_i, E);
  vals_kernel<<<(E + 255) / 256, 256, 0, stream>>>(edge_u, edge_i, deg_u, deg_i, vals, E);

  // 2) layer-sum init = layer-0 embeddings
  hipMemcpyAsync(u_sum, user_tab, NU * sizeof(float), hipMemcpyDeviceToDevice, stream);
  hipMemcpyAsync(i_sum, item_tab, NI * sizeof(float), hipMemcpyDeviceToDevice, stream);

  // 3) propagation layers (ping-pong: layer l writes ubuf[l&1] / ibuf[l&1])
  const float* ucur = user_tab;
  const float* icur = item_tab;
  float* ubufs[2] = {ubuf0, ubuf1};
  float* ibufs[2] = {ibuf0, ibuf1};
  const int spmm_grid = (int)(((size_t)E * 16 + 255) / 256);
  for (int l = 0; l < NUM_LAYER; ++l) {
    float* unxt = ubufs[l & 1];
    float* inxt = ibufs[l & 1];
    hipMemsetAsync(unxt, 0, NU * sizeof(float), stream);
    hipMemsetAsync(inxt, 0, NI * sizeof(float), stream);
    // u_nxt = A  @ i_cur : gather by edge_item, scatter to edge_user
    spmm_scatter<<<spmm_grid, 256, 0, stream>>>(icur, edge_i, edge_u, vals, unxt, E);
    // i_nxt = A^T @ u_cur : gather by edge_user, scatter to edge_item
    spmm_scatter<<<spmm_grid, 256, 0, stream>>>(ucur, edge_u, edge_i, vals, inxt, E);
    vec_add<<<(int)((NU + 255) / 256), 256, 0, stream>>>(u_sum, unxt, (int)NU);
    vec_add<<<(int)((NI + 255) / 256), 256, 0, stream>>>(i_sum, inxt, (int)NI);
    ucur = unxt;
    icur = inxt;
  }

  // 4) WMMA scoring + reg loss (1/(L+1) scale folded into gathers)
  hipMemsetAsync(reg_out, 0, sizeof(float), stream);
  score_kernel<<<BATCH / 4, 128, 0, stream>>>(u_sum, i_sum, q_user, q_item, q_negs,
                                              pos_out, neg_out, reg_out);
}
